// MultiHeadAttention_33775622815863
// MI455X (gfx1250) — matrix operand order repaired and checked by reading, not compiled, not run on hardware
//
#include <hip/hip_runtime.h>
#include <hip/hip_bf16.h>
#include <type_traits>
#include <cstdint>

// ---------------------------------------------------------------------------
// bf16 WMMA GEMM for gfx1250 (wave32), double-buffered software pipeline.
//   C = scale * A[M,K] x (B[K,N] or B[N,K]^T) (+bias), batched on blockIdx.z.
//   f32 inputs are converted to bf16 while filling LDS (sync path);
//   bf16 inputs are copied with GLOBAL_LOAD_ASYNC_TO_LDS_B128 (ASYNCcnt path),
//   with tile k+1's copies overlapped with tile k's WMMAs.
//   Block tile 128x128x32, 256 threads = 8 waves (2x4), wave tile 64x32
//   = 4x2 v_wmma_f32_16x16x32_bf16 tiles, fp32 accumulators.
//   OUT_TRANS stores C^T so consumers always see K-contiguous bf16 operands.
// ---------------------------------------------------------------------------

typedef __attribute__((ext_vector_type(16))) __bf16 v16bf;
typedef __attribute__((ext_vector_type(8)))  __bf16 v8bf;
typedef __attribute__((ext_vector_type(4)))  __bf16 v4bf;
typedef __attribute__((ext_vector_type(8)))  float  v8f;
typedef __attribute__((ext_vector_type(4)))  float  v4f;

#define BM 128
#define BN 128
#define BK 32
#define LDSP (BK + 8)   // 40 elements -> 80B row stride: 16B aligned, bank-conflict free

__device__ __forceinline__ v16bf cat16(v8bf lo, v8bf hi) {
    return __builtin_shufflevector(lo, hi, 0,1,2,3,4,5,6,7,8,9,10,11,12,13,14,15);
}

// Async global->LDS 16B copy (GVS mode: saddr base + 32-bit byte offset).
// Generic pointers into LDS carry the LDS byte offset in their low 32 bits
// (ISA 10.2 aperture mapping), so truncation yields the VDST address.
__device__ __forceinline__ void async_copy_b128(const void* gbase, uint32_t byteOff,
                                                void* ldsDst) {
    uint32_t l = (uint32_t)(uintptr_t)ldsDst;
    asm volatile("global_load_async_to_lds_b128 %0, %1, %2"
                 :: "v"(l), "v"(byteOff), "s"(gbase) : "memory");
}
__device__ __forceinline__ void wait_asynccnt0() {
    asm volatile("s_wait_asynccnt 0x0" ::: "memory");
}

template<typename TA, typename TB, bool TRANS_B, bool OUT_BF16, bool OUT_TRANS,
         bool ADD_BIAS>
__global__ __launch_bounds__(256)
void gemm_wmma(const TA* __restrict__ Ag, const TB* __restrict__ Bg,
               void* __restrict__ Cg, const float* __restrict__ bias,
               int M, int N, int K, int lda, int ldb, int ldc,
               long strideA, long strideB, long strideC, float scale)
{
    constexpr bool ASYNC_A = std::is_same<TA, __bf16>::value;
    constexpr bool ASYNC_B = std::is_same<TB, __bf16>::value && TRANS_B;

    __shared__ __bf16 As[2][BM][LDSP];   // As[buf][m][k]
    __shared__ __bf16 Bs[2][BN][LDSP];   // Bs[buf][n][k]  (K-contiguous)

    const int tid   = threadIdx.x;
    const int lane  = tid & 31;
    const int wave  = tid >> 5;
    const int waveM = wave >> 2;      // 0..1  (64 rows each)
    const int waveN = wave & 3;       // 0..3  (32 cols each)
    const int khalf = lane >> 4;      // lane half selects K/M sub-ranges
    const int ln16  = lane & 15;

    const int bm0 = blockIdx.y * BM;
    const int bn0 = blockIdx.x * BN;

    const TA* A = Ag + (long)blockIdx.z * strideA;
    const TB* B = Bg + (long)blockIdx.z * strideB;

    v8f acc[4][2];
    #pragma unroll
    for (int i = 0; i < 4; ++i)
        #pragma unroll
        for (int j = 0; j < 2; ++j)
            #pragma unroll
            for (int r = 0; r < 8; ++r) acc[i][j][r] = 0.0f;

    // ---- tile loader: global -> LDS buffer `buf` for k-offset k0 ----
    auto load_tiles = [&](int buf, int k0) {
        // A tile: 128x32
        if constexpr (ASYNC_A) {
            #pragma unroll
            for (int i = 0; i < 2; ++i) {
                int idx = tid + i * 256;            // 0..511 x 16B chunks
                int row = idx >> 2;
                int c8  = (idx & 3) * 8;
                uint32_t go = (uint32_t)(((long)(bm0 + row) * lda + k0 + c8) * 2L);
                async_copy_b128(A, go, &As[buf][row][c8]);
            }
        } else {
            #pragma unroll
            for (int i = 0; i < 4; ++i) {
                int idx = tid + i * 256;
                int row = idx >> 3;
                int c4  = (idx & 7) * 4;
                v4f v = *(const v4f*)(A + (long)(bm0 + row) * lda + k0 + c4);
                v4bf h;
                h.x = (__bf16)v.x; h.y = (__bf16)v.y;
                h.z = (__bf16)v.z; h.w = (__bf16)v.w;
                *(v4bf*)&As[buf][row][c4] = h;
            }
        }
        // B tile -> Bs[n][k]
        if constexpr (ASYNC_B) {
            #pragma unroll
            for (int i = 0; i < 2; ++i) {
                int idx = tid + i * 256;
                int n   = idx >> 2;
                int c8  = (idx & 3) * 8;
                uint32_t go = (uint32_t)(((long)(bn0 + n) * ldb + k0 + c8) * 2L);
                async_copy_b128(B, go, &Bs[buf][n][c8]);
            }
        } else if constexpr (TRANS_B) {
            // f32 [N,K] row-major: contiguous along K, convert
            #pragma unroll
            for (int i = 0; i < 4; ++i) {
                int idx = tid + i * 256;
                int n   = idx >> 3;
                int c4  = (idx & 7) * 4;
                v4f v = *(const v4f*)(B + (long)(bn0 + n) * ldb + k0 + c4);
                v4bf h;
                h.x = (__bf16)v.x; h.y = (__bf16)v.y;
                h.z = (__bf16)v.z; h.w = (__bf16)v.w;
                *(v4bf*)&Bs[buf][n][c4] = h;
            }
        } else {
            // f32 [K,N] row-major: contiguous along N, transpose into LDS
            #pragma unroll
            for (int i = 0; i < 4; ++i) {
                int idx = tid + i * 256;
                int r   = idx >> 5;
                int c4  = (idx & 31) * 4;
                v4f v = *(const v4f*)(B + (long)(k0 + r) * ldb + bn0 + c4);
                Bs[buf][c4 + 0][r] = (__bf16)v.x;
                Bs[buf][c4 + 1][r] = (__bf16)v.y;
                Bs[buf][c4 + 2][r] = (__bf16)v.z;
                Bs[buf][c4 + 3][r] = (__bf16)v.w;
            }
        }
    };

    const int steps = K / BK;
    load_tiles(0, 0);                        // prologue: fill buffer 0

    for (int s = 0; s < steps; ++s) {
        const int cur = s & 1;

        if constexpr (ASYNC_A || ASYNC_B) wait_asynccnt0();
        __syncthreads();                     // buf[cur] ready; buf[cur^1] drained

        if (s + 1 < steps)                   // overlap next tile's fetch
            load_tiles(cur ^ 1, (s + 1) * BK);

        // ---- fragments (ISA 7.12.2 layouts) + 8 WMMAs on buf[cur] ----
        v16bf afrag[4], bfrag[2];
        #pragma unroll
        for (int tm = 0; tm < 4; ++tm) {
            int row = waveM * 64 + tm * 16 + ln16;                      // M = lane
            v8bf lo = *(const v8bf*)&As[cur][row][khalf * 8];           // K 0-7 / 8-15
            v8bf hi = *(const v8bf*)&As[cur][row][16 + khalf * 8];      // K 16-23 / 24-31
            afrag[tm] = cat16(lo, hi);
        }
        #pragma unroll
        for (int tn = 0; tn < 2; ++tn) {
            int col = waveN * 32 + tn * 16 + ln16;                      // N = lane
            v8bf lo = *(const v8bf*)&Bs[cur][col][khalf * 8];           // K 0-7 / 8-15
            v8bf hi = *(const v8bf*)&Bs[cur][col][16 + khalf * 8];      // K 16-23 / 24-31
            bfrag[tn] = cat16(lo, hi);
        }
        #pragma unroll
        for (int tm = 0; tm < 4; ++tm)
            #pragma unroll
            for (int tn = 0; tn < 2; ++tn)
                acc[tm][tn] = __builtin_amdgcn_wmma_f32_16x16x32_bf16(
                    false, afrag[tm], false, bfrag[tn],
                    (short)0, acc[tm][tn], false, false);
    }

    // ---- epilogue ----
    #pragma unroll
    for (int tm = 0; tm < 4; ++tm) {
        int rowBase = bm0 + waveM * 64 + tm * 16 + khalf * 8;
        #pragma unroll
        for (int tn = 0; tn < 2; ++tn) {
            int col = bn0 + waveN * 32 + tn * 16 + ln16;
            if constexpr (OUT_BF16 && OUT_TRANS) {
                // store C^T: lane's 8 rows are contiguous -> one 16B store
                v8bf pack;
                #pragma unroll
                for (int r = 0; r < 8; ++r)
                    pack[r] = (__bf16)(acc[tm][tn][r] * scale);
                __bf16* Cb = ((__bf16*)Cg) + (long)blockIdx.z * strideC;
                *(v8bf*)(Cb + (long)col * ldc + rowBase) = pack;
            } else {
                #pragma unroll
                for (int r = 0; r < 8; ++r) {
                    long off = (long)(rowBase + r) * ldc + col;
                    float v = acc[tm][tn][r] * scale;
                    if constexpr (OUT_BF16) {
                        (((__bf16*)Cg) + (long)blockIdx.z * strideC)[off] = (__bf16)v;
                    } else {
                        if constexpr (ADD_BIAS) v += bias[off];
                        (((float*)Cg) + (long)blockIdx.z * strideC)[off] = v;
                    }
                }
            }
        }
    }
}

// ---------------------------------------------------------------------------
// In-place row softmax on bf16 matrix of row length 2048. One block per row.
// ---------------------------------------------------------------------------
__global__ __launch_bounds__(256)
void softmax_rows(__bf16* __restrict__ P, int rowLen)
{
    __shared__ float red[256];
    const int tid = threadIdx.x;
    __bf16* row = P + (long)blockIdx.x * rowLen;

    float v[8];
    float m = -3.402823466e38f;
    #pragma unroll
    for (int j = 0; j < 8; ++j) {
        v[j] = (float)row[tid + j * 256];
        m = fmaxf(m, v[j]);
    }
    red[tid] = m;
    __syncthreads();
    #pragma unroll
    for (int s = 128; s > 0; s >>= 1) {
        if (tid < s) red[tid] = fmaxf(red[tid], red[tid + s]);
        __syncthreads();
    }
    m = red[0];
    __syncthreads();

    float sum = 0.0f;
    #pragma unroll
    for (int j = 0; j < 8; ++j) {
        v[j] = __expf(v[j] - m);
        sum += v[j];
    }
    red[tid] = sum;
    __syncthreads();
    #pragma unroll
    for (int s = 128; s > 0; s >>= 1) {
        if (tid < s) red[tid] += red[tid + s];
        __syncthreads();
    }
    float inv = 1.0f / red[0];
    #pragma unroll
    for (int j = 0; j < 8; ++j)
        row[tid + j * 256] = (__bf16)(v[j] * inv);
}

// ---------------------------------------------------------------------------
// Orchestration:
//   1) Qb,Kb = data@Wq/Wk (bf16, row-major);  Vt = (data@Wv)^T  [E,S] per head
//   2) Pb = scaled Q K^T (fully async tiles)
//   3) softmax rows in place
//   4) Zt = (attn @ V)^T stored [E, H*S]  (fully async tiles)
//   5) out = W @ Z + b  (A: f32 convert, B: async from Zt)
// ---------------------------------------------------------------------------
extern "C" void kernel_launch(void* const* d_in, const int* in_sizes, int n_in,
                              void* d_out, int out_size, void* d_ws, size_t ws_size,
                              hipStream_t stream)
{
    (void)in_sizes; (void)n_in; (void)out_size; (void)ws_size;
    const int H = 16, S = 2048, D = 1024, E = 1024;

    const float* data = (const float*)d_in[0];
    const float* Wq   = (const float*)d_in[1];
    const float* Wk   = (const float*)d_in[2];
    const float* Wv   = (const float*)d_in[3];
    const float* W    = (const float*)d_in[4];
    const float* b    = (const float*)d_in[5];
    float* out = (float*)d_out;

    char* ws = (char*)d_ws;
    __bf16* Qb = (__bf16*)ws;                                     // [H,S,E]
    __bf16* Kb = Qb + (size_t)H * S * E;                          // [H,S,E]
    __bf16* Vt = Kb + (size_t)H * S * E;                          // [H,E,S]  (V^T)
    __bf16* Pb = Vt + (size_t)H * S * E;                          // [H,S,S]  scores/attn
    __bf16* Zt = Pb + (size_t)H * S * S;                          // [E,H*S]  (Z^T)

    dim3 blk(256);

    // 1) projections (f32 x f32 -> bf16); V stored transposed
    {
        dim3 grd(E / BN, S / BM, H);
        gemm_wmma<float, float, false, true, false, false><<<grd, blk, 0, stream>>>(
            data, Wq, Qb, nullptr, S, E, D, D, E, E,
            0L, (long)D * E, (long)S * E, 1.0f);
        gemm_wmma<float, float, false, true, false, false><<<grd, blk, 0, stream>>>(
            data, Wk, Kb, nullptr, S, E, D, D, E, E,
            0L, (long)D * E, (long)S * E, 1.0f);
        gemm_wmma<float, float, false, true, true, false><<<grd, blk, 0, stream>>>(
            data, Wv, Vt, nullptr, S, E, D, D, E, /*ldc=*/S,
            0L, (long)D * E, (long)E * S, 1.0f);
    }

    // 2) scores[h] = (Q[h] @ K[h]^T) / sqrt(D) -> bf16  (fully async tiles)
    {
        dim3 grd(S / BN, S / BM, H);
        gemm_wmma<__bf16, __bf16, true, true, false, false><<<grd, blk, 0, stream>>>(
            Qb, Kb, Pb, nullptr, S, S, E, E, E, S,
            (long)S * E, (long)S * E, (long)S * S, 1.0f / 32.0f);
    }

    // 3) softmax over last axis, in place
    softmax_rows<<<dim3(H * S), blk, 0, stream>>>(Pb, S);

    // 4) Z[h] = attn[h] @ V[h], stored transposed into Zt[E, H*S]
    {
        dim3 grd(E / BN, S / BM, H);
        gemm_wmma<__bf16, __bf16, true, true, true, false><<<grd, blk, 0, stream>>>(
            Pb, Vt, Zt, nullptr, S, E, S, /*lda=*/S, /*ldb=*/S, /*ldc=*/H * S,
            (long)S * S, (long)E * S, /*strideC=*/(long)S, 1.0f);
    }

    // 5) out = W(f32) @ Z + b -> f32; B read K-contiguous from Zt (async)
    {
        dim3 grd(E / BN, S / BM, 1);
        gemm_wmma<float, __bf16, true, false, false, true><<<grd, blk, 0, stream>>>(
            W, Zt, out, b, S, E, H * S, /*lda=*/H * S, /*ldb=*/H * S, /*ldc=*/E,
            0L, 0L, 0L, 1.0f);
    }
}
